// GatBlock_44727789421271
// MI455X (gfx1250) — matrix operand order repaired and checked by reading, not compiled, not run on hardware
//
#include <hip/hip_runtime.h>
#include <hip/hip_bf16.h>
#include <stdint.h>

// GAT transformer block for MI455X (gfx1250, wave32).
// GEMMs use v_wmma_f32_16x16x32_bf16 with split-bf16 (hi/lo) inputs and f32
// accumulation (~fp32 accuracy, 8x MACs/instr vs the f32 16x16x4 WMMA).
// B is staged in LDS in *fragment-major* layout so each lane fetches its
// 16-value WMMA B-fragment with two ds_load_b128 (no scalar u16 gathers).
// Graph phase uses L2-resident atomics (segment max via monotonic-u32
// encoding, segment sum, and wave-per-edge scatter-add aggregation).

typedef __attribute__((ext_vector_type(16))) __bf16 v16bf;
typedef __attribute__((ext_vector_type(8)))  float  v8f;

__device__ __forceinline__ unsigned enc_f32(float v) {
    unsigned b = __float_as_uint(v);
    return (b & 0x80000000u) ? ~b : (b | 0x80000000u);
}
__device__ __forceinline__ float dec_f32(unsigned e) {
    if (e == 0u) return 0.0f;                       // never-updated segment -> 0 (matches where(isfinite,m,0))
    unsigned b = (e & 0x80000000u) ? (e & 0x7FFFFFFFu) : ~e;
    return __uint_as_float(b);
}

// ---------------------------------------------------------------------------
// WMMA GEMM: C[M,Ntot] = epilogue(A[M,K] @ B[K,Ntot])
// Block: 256 threads = 8 waves, each wave owns a 16-row x NB-col tile slice.
// B tile (K x NB) staged in LDS as bf16 hi+lo (64KB total), fragment-major:
//   slot fl = (kb*NBT + nb)*32 + lane holds that lane's 16 contiguous bf16.
// EPI: 0 = plain, 1 = relu(+bias), 2 = +bias +residual
// ---------------------------------------------------------------------------
template <int K, int NB, int EPI>
__global__ __launch_bounds__(256) void gemm_wmma_kernel(
    const float* __restrict__ A, const float* __restrict__ B,
    const float* __restrict__ bias, const float* __restrict__ res,
    float* __restrict__ C, int M, int Ntot)
{
    constexpr int KB   = K / 32;        // K-slabs of 32
    constexpr int NBT  = NB / 16;       // 16-col tiles per block
    constexpr int NFRG = KB * NBT * 32; // fragment-lane slots
    __shared__ __align__(32) __bf16 sBhi[NFRG * 16];
    __shared__ __align__(32) __bf16 sBlo[NFRG * 16];

    const int tid  = threadIdx.x;
    const int col0 = blockIdx.y * NB;

    // Stage B chunk into LDS, swizzled to fragment-major, split hi/lo bf16.
    // Thread handles whole fragment-lanes: 16 strided global reads (coalesced
    // across the 16 adjacent-column lanes), one v16bf LDS store each.
    for (int fl = tid; fl < NFRG; fl += 256) {
        const int kb   = fl / (NBT * 32);
        const int rem  = fl % (NBT * 32);
        const int nb   = rem >> 5;
        const int ln   = rem & 31;
        const int c    = col0 + nb * 16 + (ln & 15);
        const int k0   = kb * 32 + (ln >> 4) * 16;
        v16bf hi, lo;
        #pragma unroll
        for (int j = 0; j < 16; ++j) {
            float v = B[(size_t)(k0 + j) * Ntot + c];
            __bf16 hh = (__bf16)v;
            hi[j] = hh;
            lo[j] = (__bf16)(v - (float)hh);
        }
        ((v16bf*)sBhi)[fl] = hi;
        ((v16bf*)sBlo)[fl] = lo;
    }
    __syncthreads();

    const int wave = tid >> 5, lane = tid & 31;
    const int lrow = lane & 15, half = lane >> 4;
    const int arow = blockIdx.x * 128 + wave * 16 + lrow;  // A-fragment row for this lane
    const bool rowok = arow < M;
    const float* Arow = A + (size_t)arow * K;
    const v16bf* Bhi = (const v16bf*)sBhi;
    const v16bf* Blo = (const v16bf*)sBlo;

    v8f acc[NBT] = {};

    #pragma unroll
    for (int kb = 0; kb < KB; ++kb) {
        // A fragment (16x32 bf16 layout): lane(l&15)=row, half selects K-subgroup.
        // Element order = two contiguous 8-float runs of the row.
        v16bf ah, al;
        const int o0 = kb * 32 + half * 16;      // elems 0..7  <- K o0..o0+7
        const int o1 = o0 + 8;                   // elems 8..15 <- K o1..o1+7
        #pragma unroll
        for (int j = 0; j < 8; ++j) {
            float v0 = rowok ? Arow[o0 + j] : 0.0f;
            float v1 = rowok ? Arow[o1 + j] : 0.0f;
            __bf16 h0 = (__bf16)v0, h1 = (__bf16)v1;
            ah[j]     = h0;
            ah[8 + j] = h1;
            al[j]     = (__bf16)(v0 - (float)h0);
            al[8 + j] = (__bf16)(v1 - (float)h1);
        }
        #pragma unroll
        for (int nb = 0; nb < NBT; ++nb) {
            const int fl = (kb * NBT + nb) * 32 + lane;
            v16bf bh = Bhi[fl];   // 2x ds_load_b128
            v16bf bl = Blo[fl];
            acc[nb] = __builtin_amdgcn_wmma_f32_16x16x32_bf16(false, ah, false, bh, (short)0, acc[nb], false, false);
            acc[nb] = __builtin_amdgcn_wmma_f32_16x16x32_bf16(false, al, false, bh, (short)0, acc[nb], false, false);
            acc[nb] = __builtin_amdgcn_wmma_f32_16x16x32_bf16(false, ah, false, bl, (short)0, acc[nb], false, false);
        }
    }

    // Epilogue. C layout: lanes 0-15 rows r, lanes 16-31 rows r+8; col = lane&15.
    #pragma unroll
    for (int nb = 0; nb < NBT; ++nb) {
        const int cc = col0 + nb * 16 + lrow;
        #pragma unroll
        for (int r = 0; r < 8; ++r) {
            int rr = blockIdx.x * 128 + wave * 16 + r + 8 * half;
            if (rr < M) {
                float v = acc[nb][r];
                if (EPI == 1) { v += bias[cc]; v = fmaxf(v, 0.0f); }
                if (EPI == 2) { v += bias[cc] + res[(size_t)rr * Ntot + cc]; }
                C[(size_t)rr * Ntot + cc] = v;
            }
        }
    }
}

// ---------------------------------------------------------------------------
// Zero-init accumulators (g, denom) and the encoded-max buffer.
// ---------------------------------------------------------------------------
__global__ void init_kernel(float* __restrict__ g, float* __restrict__ denom,
                            unsigned* __restrict__ menc, int n)
{
    int i = blockIdx.x * blockDim.x + threadIdx.x;
    if (i < n * 128) g[i] = 0.0f;
    if (i < n * 8) { denom[i] = 0.0f; menc[i] = 0u; }
}

// a_src[n,h] = <h[n,h,:], att_src[h,:]>, same for dst. One thread per (n,h).
__global__ void attn_coef_kernel(const float* __restrict__ h,
                                 const float* __restrict__ asv,
                                 const float* __restrict__ adv,
                                 float* __restrict__ out_s,
                                 float* __restrict__ out_d, int n)
{
    int gid = blockIdx.x * blockDim.x + threadIdx.x;
    int node = gid >> 3, hd = gid & 7;
    if (node >= n) return;
    float s1 = 0.0f, s2 = 0.0f;
    #pragma unroll
    for (int f = 0; f < 16; ++f) {
        float v = h[(size_t)node * 128 + hd * 16 + f];
        s1 += v * asv[hd * 16 + f];
        s2 += v * adv[hd * 16 + f];
    }
    out_s[(size_t)node * 8 + hd] = s1;
    out_d[(size_t)node * 8 + hd] = s2;
}

// Pass 1: e = leaky_relu(a_src[src]+a_dst[dst]); segment max via u32 atomicMax.
__global__ void edge_max_kernel(const long long* __restrict__ ei,
                                const float* __restrict__ as,
                                const float* __restrict__ ad,
                                float* __restrict__ e,
                                unsigned* __restrict__ menc, int E, int Etot)
{
    int i = blockIdx.x * blockDim.x + threadIdx.x;
    if (i >= Etot) return;
    int s, d;
    if (i < E) { s = (int)ei[i]; d = (int)ei[(size_t)E + i]; }
    else       { s = d = i - E; }                       // self loops appended
    #pragma unroll
    for (int hd = 0; hd < 8; ++hd) {
        float v = as[(size_t)s * 8 + hd] + ad[(size_t)d * 8 + hd];
        v = v > 0.0f ? v : 0.2f * v;                    // leaky relu, slope 0.2
        e[(size_t)i * 8 + hd] = v;
        atomicMax(&menc[(size_t)d * 8 + hd], enc_f32(v));
    }
}

// Pass 2: p = exp(e - m[dst]); denom[dst] += p (p overwrites e).
__global__ void edge_sum_kernel(const long long* __restrict__ ei,
                                const unsigned* __restrict__ menc,
                                float* __restrict__ e,
                                float* __restrict__ denom, int E, int Etot)
{
    int i = blockIdx.x * blockDim.x + threadIdx.x;
    if (i >= Etot) return;
    int d = (i < E) ? (int)ei[(size_t)E + i] : (i - E);
    #pragma unroll
    for (int hd = 0; hd < 8; ++hd) {
        float m = dec_f32(menc[(size_t)d * 8 + hd]);
        float p = expf(e[(size_t)i * 8 + hd] - m);
        e[(size_t)i * 8 + hd] = p;
        atomicAdd(&denom[(size_t)d * 8 + hd], p);
    }
}

// Pass 3: one wave per edge; lane covers 4 consecutive features of one head.
// g[dst] += alpha * h[src]   (alpha = p / (denom[dst] + 1e-16))
__global__ void edge_aggr_kernel(const long long* __restrict__ ei,
                                 const float* __restrict__ p,
                                 const float* __restrict__ denom,
                                 const float* __restrict__ h,
                                 float* __restrict__ g, int E, int Etot)
{
    int gid = blockIdx.x * blockDim.x + threadIdx.x;
    int i = gid >> 5, lane = gid & 31;
    if (i >= Etot) return;
    int s, d;
    if (i < E) { s = (int)ei[i]; d = (int)ei[(size_t)E + i]; }
    else       { s = d = i - E; }
    const int hd = lane >> 2, f0 = (lane & 3) * 4;
    const float alpha = p[(size_t)i * 8 + hd] /
                        (denom[(size_t)d * 8 + hd] + 1e-16f);
    const float4 hv = *(const float4*)(h + (size_t)s * 128 + hd * 16 + f0);
    float* go = g + (size_t)d * 128 + hd * 16 + f0;
    atomicAdd(go + 0, alpha * hv.x);
    atomicAdd(go + 1, alpha * hv.y);
    atomicAdd(go + 2, alpha * hv.z);
    atomicAdd(go + 3, alpha * hv.w);
}

// LayerNorm over D=128: one wave per node, 4 elems/lane, shfl_xor reductions.
// out = LN(a (+ b) (+ bias)) * gamma + beta
__global__ void ln_kernel(const float* __restrict__ a, const float* __restrict__ b,
                          const float* __restrict__ bias,
                          const float* __restrict__ gamma,
                          const float* __restrict__ beta,
                          float* __restrict__ out, int n)
{
    int gid = blockIdx.x * blockDim.x + threadIdx.x;
    int node = gid >> 5, lane = gid & 31;
    if (node >= n) return;
    float v[4];
    float s = 0.0f;
    #pragma unroll
    for (int j = 0; j < 4; ++j) {
        int f = lane * 4 + j;
        float x = a[(size_t)node * 128 + f];
        if (b)    x += b[(size_t)node * 128 + f];
        if (bias) x += bias[f];
        v[j] = x;
        s += x;
    }
    #pragma unroll
    for (int m = 16; m >= 1; m >>= 1) s += __shfl_xor(s, m, 32);
    const float mu = s * (1.0f / 128.0f);
    float q = 0.0f;
    #pragma unroll
    for (int j = 0; j < 4; ++j) { float dd = v[j] - mu; q += dd * dd; }
    #pragma unroll
    for (int m = 16; m >= 1; m >>= 1) q += __shfl_xor(q, m, 32);
    const float rs = rsqrtf(q * (1.0f / 128.0f) + 1e-5f);
    #pragma unroll
    for (int j = 0; j < 4; ++j) {
        int f = lane * 4 + j;
        out[(size_t)node * 128 + f] = (v[j] - mu) * rs * gamma[f] + beta[f];
    }
}

// ---------------------------------------------------------------------------
extern "C" void kernel_launch(void* const* d_in, const int* in_sizes, int n_in,
                              void* d_out, int out_size, void* d_ws, size_t ws_size,
                              hipStream_t stream)
{
    const float*     x       = (const float*)d_in[0];
    const long long* ei      = (const long long*)d_in[1];   // int64 edge_index [2,E]
    const float*     W_gat   = (const float*)d_in[2];
    const float*     att_src = (const float*)d_in[3];
    const float*     att_dst = (const float*)d_in[4];
    const float*     bias_g  = (const float*)d_in[5];
    const float*     gamma1  = (const float*)d_in[6];
    const float*     beta1   = (const float*)d_in[7];
    const float*     w_ff1   = (const float*)d_in[8];
    const float*     b_ff1   = (const float*)d_in[9];
    const float*     w_ff2   = (const float*)d_in[10];
    const float*     b_ff2   = (const float*)d_in[11];
    const float*     gamma2  = (const float*)d_in[12];
    const float*     beta2   = (const float*)d_in[13];
    float* zout = (float*)d_out;

    const int N    = in_sizes[0] / 128;   // 50000 nodes
    const int E    = in_sizes[1] / 2;     // 800000 edges
    const int Etot = E + N;               // + self loops

    // Workspace layout (t aliases the dead h..e region; y aliases g).
    char* ws = (char*)d_ws;
    size_t off = 0;
    auto alloc = [&](size_t bytes) { size_t o = off; off += (bytes + 255) & ~(size_t)255; return o; };
    const size_t o_h   = alloc((size_t)N * 128 * 4);
    const size_t o_as  = alloc((size_t)N * 8 * 4);
    const size_t o_ad  = alloc((size_t)N * 8 * 4);
    const size_t o_m   = alloc((size_t)N * 8 * 4);
    const size_t o_den = alloc((size_t)N * 8 * 4);
    const size_t o_e   = alloc((size_t)Etot * 8 * 4);
    size_t o_t = o_h;                                   // FFN hidden aliases region A
    if (off < (size_t)N * 256 * 4) o_t = alloc((size_t)N * 256 * 4);
    const size_t o_g = alloc((size_t)N * 128 * 4);
    const size_t o_y = o_g;                             // y aliases g (dead after LN1)
    const size_t o_u = alloc((size_t)N * 128 * 4);

    float*    h    = (float*)(ws + o_h);
    float*    a_s  = (float*)(ws + o_as);
    float*    a_d  = (float*)(ws + o_ad);
    unsigned* menc = (unsigned*)(ws + o_m);
    float*    den  = (float*)(ws + o_den);
    float*    e    = (float*)(ws + o_e);
    float*    t    = (float*)(ws + o_t);
    float*    g    = (float*)(ws + o_g);
    float*    y    = (float*)(ws + o_y);
    float*    u    = (float*)(ws + o_u);

    const int rowTiles = (N + 127) / 128;

    // 0) zero accumulators
    init_kernel<<<(N * 128 + 255) / 256, 256, 0, stream>>>(g, den, menc, N);
    // 1) h = x @ W_gat  (WMMA bf16 split-precision)
    gemm_wmma_kernel<128, 128, 0><<<dim3(rowTiles, 1), 256, 0, stream>>>(
        x, W_gat, nullptr, nullptr, h, N, 128);
    // 2) per-(node,head) attention coefficients
    attn_coef_kernel<<<(N * 8 + 255) / 256, 256, 0, stream>>>(h, att_src, att_dst, a_s, a_d, N);
    // 3) edge logits + segment max
    edge_max_kernel<<<(Etot + 255) / 256, 256, 0, stream>>>(ei, a_s, a_d, e, menc, E, Etot);
    // 4) exp + segment sum
    edge_sum_kernel<<<(Etot + 255) / 256, 256, 0, stream>>>(ei, menc, e, den, E, Etot);
    // 5) weighted scatter-add aggregation (wave per edge)
    edge_aggr_kernel<<<(Etot * 32 + 255) / 256, 256, 0, stream>>>(ei, e, den, h, g, E, Etot);
    // 6) u = LN(x + g + bias_gat)
    ln_kernel<<<(N * 32 + 255) / 256, 256, 0, stream>>>(x, g, bias_g, gamma1, beta1, u, N);
    // 7) t = relu(u @ w_ff1 + b_ff1)
    gemm_wmma_kernel<128, 128, 1><<<dim3(rowTiles, 2), 256, 0, stream>>>(
        u, w_ff1, b_ff1, nullptr, t, N, 256);
    // 8) y = u + t @ w_ff2 + b_ff2
    gemm_wmma_kernel<256, 64, 2><<<dim3(rowTiles, 2), 256, 0, stream>>>(
        t, w_ff2, b_ff2, u, y, N, 128);
    // 9) z = LN(y)
    ln_kernel<<<(N * 32 + 255) / 256, 256, 0, stream>>>(y, nullptr, nullptr, gamma2, beta2, zout, N);
}
